// DiacriticRestorer_4578435137754
// MI455X (gfx1250) — compile-verified, matrix operand order checked by reading, not run on hardware
//
#include <hip/hip_runtime.h>
#include <stdint.h>

// ---------------- problem constants ----------------
#define BATCH 64
#define SEQT  512
#define EDIM  256
#define HDIM  512
#define VDIM  128
#define G4    (4 * HDIM)          // 2048
#define BT    (BATCH * SEQT)      // 32768

// ---------------- WMMA types ----------------
typedef __attribute__((ext_vector_type(16))) __bf16 bf16x16;
typedef __attribute__((ext_vector_type(8)))  float  f32x8;

union FragAB {
  bf16x16 v;
  uint4   q[2];
};

static __device__ __forceinline__ f32x8 wmma_bf16(bf16x16 a, bf16x16 b, f32x8 c) {
  // (neg_a, A, neg_b, B, c_mod, C, reuse_a, reuse_b)
  return __builtin_amdgcn_wmma_f32_16x16x32_bf16(false, a, false, b, (short)0, c, false, false);
}

// f32 -> bf16 (round to nearest even)
static __device__ __forceinline__ unsigned short f2bf(float x) {
  unsigned u = __float_as_uint(x);
  unsigned r = u + 0x7FFFu + ((u >> 16) & 1u);
  return (unsigned short)(r >> 16);
}

static __device__ __forceinline__ float sigf(float x) {
  return 1.0f / (1.0f + __expf(-x));
}

// Load a 16x32 (A) or 32x16-as-[N][K] (B) bf16 fragment from row-major storage.
// lane L: row = base_row + (L & 15); K chunks at k + (L>>4)*8 and k + 16 + (L>>4)*8.
static __device__ __forceinline__ bf16x16 load_frag(const unsigned short* __restrict__ base,
                                                    size_t ld, int row, int k, int lane) {
  const unsigned short* p = base + (size_t)(row + (lane & 15)) * ld
                                 + (size_t)k + (size_t)((lane >> 4) << 3);
  FragAB f;
  f.q[0] = *(const uint4*)(p);
  f.q[1] = *(const uint4*)(p + 16);
  return f.v;
}

// ---------------- elementwise helper kernels ----------------
__global__ void k_cvt_bf16(const float* __restrict__ src, unsigned short* __restrict__ dst, int n) {
  int i = blockIdx.x * blockDim.x + threadIdx.x;
  if (i < n) dst[i] = f2bf(src[i]);
}

__global__ void k_bias_sum(const float* __restrict__ a, const float* __restrict__ b,
                           float* __restrict__ o, int n) {
  int i = blockIdx.x * blockDim.x + threadIdx.x;
  if (i < n) o[i] = a[i] + b[i];
}

// Embedding gather -> bf16, written into dec_in[:, :, 0:EDIM] (row stride ldd)
__global__ void k_embed(const int* __restrict__ x, const float* __restrict__ emb,
                        unsigned short* __restrict__ dst, int ldd) {
  int i = blockIdx.x * blockDim.x + threadIdx.x;   // over BT*EDIM
  if (i >= BT * EDIM) return;
  int e  = i % EDIM;
  int bt = i / EDIM;
  int tok = x[bt];
  dst[(size_t)bt * ldd + e] = f2bf(emb[(size_t)tok * EDIM + e]);
}

// ---------------- big GEMM: out[M][N] = A[M][K](bf16) * W[N][K]^T(bf16) + bias ----------------
// block = 256 thr = 8 waves; block tile 64(M) x 128(N); wave tile 32x32 (2x2 frags)
__global__ __launch_bounds__(256)
void k_gemm_bf16(const unsigned short* __restrict__ A, int ldA,
                 const unsigned short* __restrict__ W,
                 const float* __restrict__ bias,
                 float* __restrict__ out, int ldOut,
                 int M, int N, int K) {
  int lane = threadIdx.x & 31;
  int wave = threadIdx.x >> 5;        // 0..7
  int wm = wave >> 2;                 // 0..1
  int wn = wave & 3;                  // 0..3
  int bm = blockIdx.x * 64 + wm * 32;
  int bn = blockIdx.y * 128 + wn * 32;
  (void)M;

  f32x8 acc[2][2] = {};
  for (int k = 0; k < K; k += 32) {
    if (k + 32 < K)
      __builtin_prefetch(W + (size_t)(bn + (lane & 15)) * K + k + 32, 0, 1);
    bf16x16 a0 = load_frag(A, (size_t)ldA, bm,      k, lane);
    bf16x16 a1 = load_frag(A, (size_t)ldA, bm + 16, k, lane);
    bf16x16 b0 = load_frag(W, (size_t)K,   bn,      k, lane);
    bf16x16 b1 = load_frag(W, (size_t)K,   bn + 16, k, lane);
    acc[0][0] = wmma_bf16(a0, b0, acc[0][0]);
    acc[0][1] = wmma_bf16(a0, b1, acc[0][1]);
    acc[1][0] = wmma_bf16(a1, b0, acc[1][0]);
    acc[1][1] = wmma_bf16(a1, b1, acc[1][1]);
  }

  int rsub = (lane >> 4) << 3;        // 0 or 8
#pragma unroll
  for (int mt = 0; mt < 2; ++mt)
#pragma unroll
    for (int nt = 0; nt < 2; ++nt) {
      int col = bn + nt * 16 + (lane & 15);
      float bv = bias ? bias[col] : 0.0f;
      int rbase = bm + mt * 16 + rsub;
#pragma unroll
      for (int r = 0; r < 8; ++r)
        out[(size_t)(rbase + r) * ldOut + col] = acc[mt][nt][r] + bv;
    }
}

// ---------------- persistent LSTM scan ----------------
// Grid: 4 blocks (16 batch rows each) x 512 threads (16 waves).
// wave w owns hidden slice j in [w*32, w*32+32): computes gate columns
// {g*H + jw + hh*16} for g=0..3, hh=0..1 -> 8 accumulator fragments.
// h lives in LDS (bf16); c lives in registers in the D-fragment lane mapping.
#define BB   16
#define LDH  (HDIM + 8)

__global__ __launch_bounds__(512)
void k_lstm_scan(const float* __restrict__ xg,              // [B][T][4H] f32 (bias included)
                 const unsigned short* __restrict__ Whh,    // [4H][H] bf16
                 const unsigned short* __restrict__ initH,  // [B][H] bf16 or null
                 const float* __restrict__ initC,           // [B][H] f32 or null
                 unsigned short* __restrict__ y, int ldy, int yofs,
                 unsigned short* __restrict__ hT,           // [B][H] bf16 or null
                 float* __restrict__ cT,                    // [B][H] f32 or null
                 int reverse) {
  __shared__ __align__(16) unsigned short hbuf[BB][LDH];

  int lane = threadIdx.x & 31;
  int wave = threadIdx.x >> 5;        // 0..15
  int b0 = blockIdx.x * BB;
  int jw = wave * 32;
  int rrow = (lane >> 4) << 3;        // 0 or 8  (batch sub-row)
  int lcol = lane & 15;               // hidden column within 16-tile

  // ---- init h (LDS) and c (registers) ----
  for (int i = threadIdx.x; i < BB * HDIM; i += blockDim.x) {
    int b = i / HDIM, j = i % HDIM;
    hbuf[b][j] = initH ? initH[(size_t)(b0 + b) * HDIM + j] : (unsigned short)0;
  }
  float creg[2][8];
#pragma unroll
  for (int hh = 0; hh < 2; ++hh) {
    int j = jw + hh * 16 + lcol;
#pragma unroll
    for (int r = 0; r < 8; ++r)
      creg[hh][r] = initC ? initC[(size_t)(b0 + rrow + r) * HDIM + j] : 0.0f;
  }
  __syncthreads();

  for (int t = 0; t < SEQT; ++t) {
    int te = reverse ? (SEQT - 1 - t) : t;

    // ---- load gate pre-activations as C initializers ----
    f32x8 acc[4][2];
#pragma unroll
    for (int g = 0; g < 4; ++g)
#pragma unroll
      for (int hh = 0; hh < 2; ++hh) {
        int col = g * HDIM + jw + hh * 16 + lcol;
        const float* xp = xg + ((size_t)(b0 + rrow) * SEQT + te) * G4 + col;
        f32x8 v;
#pragma unroll
        for (int r = 0; r < 8; ++r)
          v[r] = xp[(size_t)r * SEQT * G4];
        acc[g][hh] = v;
      }

    // ---- gates += h @ Whh^T over the wave's 128 gate columns ----
    for (int k = 0; k < HDIM; k += 32) {
      bf16x16 a = load_frag(&hbuf[0][0], LDH, 0, k, lane);
      if (k + 32 < HDIM)
        __builtin_prefetch(Whh + (size_t)(jw + (lane & 15)) * HDIM + k + 32, 0, 1);
#pragma unroll
      for (int g = 0; g < 4; ++g)
#pragma unroll
        for (int hh = 0; hh < 2; ++hh) {
          bf16x16 b = load_frag(Whh, (size_t)HDIM, g * HDIM + jw + hh * 16, k, lane);
          acc[g][hh] = wmma_bf16(a, b, acc[g][hh]);
        }
    }
    __syncthreads();   // all waves done reading hbuf

    // ---- elementwise LSTM cell update (all gates resident in this lane) ----
#pragma unroll
    for (int hh = 0; hh < 2; ++hh) {
      int j = jw + hh * 16 + lcol;
#pragma unroll
      for (int r = 0; r < 8; ++r) {
        int b = rrow + r;
        float iv = sigf(acc[0][hh][r]);
        float fv = sigf(acc[1][hh][r]);
        float gv = tanhf(acc[2][hh][r]);
        float ov = sigf(acc[3][hh][r]);
        float c  = fv * creg[hh][r] + iv * gv;
        creg[hh][r] = c;
        float h = ov * tanhf(c);
        unsigned short hb = f2bf(h);
        hbuf[b][j] = hb;
        y[((size_t)(b0 + b) * SEQT + te) * ldy + yofs + j] = hb;
      }
    }
    __syncthreads();   // new h visible before next step's GEMM
  }

  // ---- save final states (used for decoder init) ----
  if (hT) {
    for (int i = threadIdx.x; i < BB * HDIM; i += blockDim.x) {
      int b = i / HDIM, j = i % HDIM;
      hT[(size_t)(b0 + b) * HDIM + j] = hbuf[b][j];
    }
  }
  if (cT) {
#pragma unroll
    for (int hh = 0; hh < 2; ++hh) {
      int j = jw + hh * 16 + lcol;
#pragma unroll
      for (int r = 0; r < 8; ++r)
        cT[(size_t)(b0 + rrow + r) * HDIM + j] = creg[hh][r];
    }
  }
}

// ---------------- host orchestration ----------------
extern "C" void kernel_launch(void* const* d_in, const int* in_sizes, int n_in,
                              void* d_out, int out_size, void* d_ws, size_t ws_size,
                              hipStream_t stream) {
  (void)in_sizes; (void)n_in; (void)out_size; (void)ws_size;

  uint8_t* wsb = (uint8_t*)d_ws;
  size_t off = 0;
  auto alloc = [&](size_t bytes) -> void* {
    void* p = wsb + off;
    off = (off + bytes + 255) & ~(size_t)255;
    return p;
  };

  // scratch buffers
  float*          xg     = (float*)alloc((size_t)BT * G4 * 4);          // reused per scan
  unsigned short* x1     = (unsigned short*)alloc((size_t)BT * 2 * HDIM * 2); // [B,T,2H]
  unsigned short* dec_in = (unsigned short*)alloc((size_t)BT * (EDIM + 2 * HDIM) * 2); // [B,T,1280]
  unsigned short* yd0    = (unsigned short*)alloc((size_t)BT * HDIM * 2);
  unsigned short* yd1    = (unsigned short*)alloc((size_t)BT * HDIM * 2);

  // bf16 weights
  unsigned short* wih_l0f = (unsigned short*)alloc((size_t)G4 * EDIM * 2);
  unsigned short* wih_l0b = (unsigned short*)alloc((size_t)G4 * EDIM * 2);
  unsigned short* wih_l1f = (unsigned short*)alloc((size_t)G4 * 2 * HDIM * 2);
  unsigned short* wih_l1b = (unsigned short*)alloc((size_t)G4 * 2 * HDIM * 2);
  unsigned short* wih_d0  = (unsigned short*)alloc((size_t)G4 * (EDIM + 2 * HDIM) * 2);
  unsigned short* wih_d1  = (unsigned short*)alloc((size_t)G4 * HDIM * 2);
  unsigned short* whh[6];
  for (int i = 0; i < 6; ++i) whh[i] = (unsigned short*)alloc((size_t)G4 * HDIM * 2);
  unsigned short* wout = (unsigned short*)alloc((size_t)VDIM * HDIM * 2);
  float* bsum[6];
  for (int i = 0; i < 6; ++i) bsum[i] = (float*)alloc((size_t)G4 * 4);

  // saved encoder-l0 final states (decoder init)
  unsigned short* h0f = (unsigned short*)alloc((size_t)BATCH * HDIM * 2);
  float*          c0f = (float*)alloc((size_t)BATCH * HDIM * 4);
  unsigned short* h0b = (unsigned short*)alloc((size_t)BATCH * HDIM * 2);
  float*          c0b = (float*)alloc((size_t)BATCH * HDIM * 4);

  auto cvt = [&](int idx, unsigned short* dst, int n) {
    k_cvt_bf16<<<(n + 255) / 256, 256, 0, stream>>>((const float*)d_in[idx], dst, n);
  };
  // weight conversions (input-index layout from setup_inputs order)
  cvt(2,  wih_l0f, G4 * EDIM);
  cvt(6,  wih_l0b, G4 * EDIM);
  cvt(10, wih_l1f, G4 * 2 * HDIM);
  cvt(14, wih_l1b, G4 * 2 * HDIM);
  cvt(18, wih_d0,  G4 * (EDIM + 2 * HDIM));
  cvt(22, wih_d1,  G4 * HDIM);
  cvt(3,  whh[0], G4 * HDIM);
  cvt(7,  whh[1], G4 * HDIM);
  cvt(11, whh[2], G4 * HDIM);
  cvt(15, whh[3], G4 * HDIM);
  cvt(19, whh[4], G4 * HDIM);
  cvt(23, whh[5], G4 * HDIM);
  cvt(26, wout,   VDIM * HDIM);

  const int bpair[6][2] = {{4,5},{8,9},{12,13},{16,17},{20,21},{24,25}};
  for (int i = 0; i < 6; ++i)
    k_bias_sum<<<(G4 + 255) / 256, 256, 0, stream>>>(
        (const float*)d_in[bpair[i][0]], (const float*)d_in[bpair[i][1]], bsum[i], G4);

  // embedding -> dec_in[:, :, 0:E]
  k_embed<<<(BT * EDIM) / 256, 256, 0, stream>>>(
      (const int*)d_in[0], (const float*)d_in[1], dec_in, EDIM + 2 * HDIM);

  auto gemm = [&](const unsigned short* A, int ldA, const unsigned short* W,
                  const float* bias, float* out, int ldOut, int M, int N, int K) {
    dim3 g(M / 64, N / 128);
    k_gemm_bf16<<<g, 256, 0, stream>>>(A, ldA, W, bias, out, ldOut, M, N, K);
  };
  auto scan = [&](const unsigned short* whh_, const unsigned short* iH, const float* iC,
                  unsigned short* yp, int ldy, int yofs,
                  unsigned short* hTp, float* cTp, int rev) {
    k_lstm_scan<<<BATCH / BB, 512, 0, stream>>>(xg, whh_, iH, iC, yp, ldy, yofs, hTp, cTp, rev);
  };

  const int LDDEC = EDIM + 2 * HDIM;   // 1280

  // encoder layer 0 forward / backward (input = embedded, K=E)
  gemm(dec_in, LDDEC, wih_l0f, bsum[0], xg, G4, BT, G4, EDIM);
  scan(whh[0], nullptr, nullptr, x1, 2 * HDIM, 0,      h0f, c0f, 0);
  gemm(dec_in, LDDEC, wih_l0b, bsum[1], xg, G4, BT, G4, EDIM);
  scan(whh[1], nullptr, nullptr, x1, 2 * HDIM, HDIM,   h0b, c0b, 1);

  // encoder layer 1 forward / backward (input = x1, K=2H); y -> dec_in[:, :, E : E+2H]
  gemm(x1, 2 * HDIM, wih_l1f, bsum[2], xg, G4, BT, G4, 2 * HDIM);
  scan(whh[2], nullptr, nullptr, dec_in, LDDEC, EDIM,        nullptr, nullptr, 0);
  gemm(x1, 2 * HDIM, wih_l1b, bsum[3], xg, G4, BT, G4, 2 * HDIM);
  scan(whh[3], nullptr, nullptr, dec_in, LDDEC, EDIM + HDIM, nullptr, nullptr, 1);

  // decoder layer 0 (input = [emb | enc_out], K=1280, init = enc l0 fwd finals)
  gemm(dec_in, LDDEC, wih_d0, bsum[4], xg, G4, BT, G4, LDDEC);
  scan(whh[4], h0f, c0f, yd0, HDIM, 0, nullptr, nullptr, 0);

  // decoder layer 1 (input = yd0, K=H, init = enc l0 bwd finals)
  gemm(yd0, HDIM, wih_d1, bsum[5], xg, G4, BT, G4, HDIM);
  scan(whh[5], h0b, c0b, yd1, HDIM, 0, nullptr, nullptr, 0);

  // logits = yd1 @ out_W^T + out_b  -> d_out f32 [B,T,V]
  gemm(yd1, HDIM, wout, (const float*)d_in[27], (float*)d_out, VDIM, BT, VDIM, HDIM);
}
